// MultiHeadAttention_45552423141412
// MI455X (gfx1250) — matrix operand order repaired and checked by reading, not compiled, not run on hardware
//
#include <hip/hip_runtime.h>

// ---------------------------------------------------------------------------
// MHA forward (RoPE + causal flash attention) for gfx1250 (MI455X, wave32).
// - All matmuls on v_wmma_f32_16x16x32_f16 (f16 in / f32 accumulate).
// - V stored transposed so every WMMA B-fragment is a contiguous v16h load.
// - Flash attention double-buffers K / V^T tiles into LDS with the Tensor
//   Data Mover (tensor_load_to_lds + s_wait_tensorcnt): per-wave DMA, no
//   block barriers, DMA overlaps WMMA+softmax of the previous tile.
// - Working set (~48 MB f16) fits in the 192 MB L2.
// ---------------------------------------------------------------------------

typedef __attribute__((ext_vector_type(16))) _Float16 v16h;
typedef __attribute__((ext_vector_type(8)))  _Float16 v8h;
typedef __attribute__((ext_vector_type(8)))  float    v8f;
typedef __attribute__((ext_vector_type(4)))  unsigned int v4u;
typedef __attribute__((ext_vector_type(8)))  int      v8i;
typedef __attribute__((ext_vector_type(4)))  int      v4i;

#define D_MODEL   1024
#define NUM_HEADS 16
#define DK        64
#define SEQ       2048
#define BATCH     2
#define MTOT      (BATCH * SEQ)       // 4096 tokens
#define NEG_BIG   (-3.0e38f)

// ---------------------------------------------------------------------------
// f32 -> f16 conversion (grid-stride)
// ---------------------------------------------------------------------------
__global__ void cvt_f32_to_f16(const float* __restrict__ in,
                               _Float16* __restrict__ out, int n) {
  int i = blockIdx.x * blockDim.x + threadIdx.x;
  int stride = gridDim.x * blockDim.x;
  for (; i < n; i += stride) out[i] = (_Float16)in[i];
}

// ---------------------------------------------------------------------------
// TDM: async DMA of a 2D tile (2-byte elements) from global into LDS.
// Descriptor per CDNA5 ISA:
//   group0: [0]=count=1, [1]=lds_addr, [2]=global_addr[31:0],
//           [3]=global_addr[56:32] | type=2<<30
//   group1: [0]=data_size(2B)<<16, dims/tile/stride packed per spec
// All operands are wave-uniform -> readfirstlane to force SGPR materialization.
// Tracked with TENSORcnt (per-wave, in-order).
// ---------------------------------------------------------------------------
__device__ __forceinline__ void tdm_load_2d(unsigned lds_byte_addr,
                                            const void* gptr,
                                            unsigned tensor_d0, unsigned tensor_d1,
                                            unsigned tile_d0, unsigned tile_d1,
                                            unsigned stride0_elems) {
  unsigned long long ga = (unsigned long long)(uintptr_t)gptr;
  unsigned ga_lo = (unsigned)__builtin_amdgcn_readfirstlane((int)(unsigned)ga);
  unsigned ga_hi = (unsigned)__builtin_amdgcn_readfirstlane((int)(unsigned)(ga >> 32));
  unsigned lb    = (unsigned)__builtin_amdgcn_readfirstlane((int)lds_byte_addr);
  unsigned td0   = (unsigned)__builtin_amdgcn_readfirstlane((int)tensor_d0);
  unsigned td1   = (unsigned)__builtin_amdgcn_readfirstlane((int)tensor_d1);
  unsigned tl0   = (unsigned)__builtin_amdgcn_readfirstlane((int)tile_d0);
  unsigned tl1   = (unsigned)__builtin_amdgcn_readfirstlane((int)tile_d1);
  unsigned st0   = (unsigned)__builtin_amdgcn_readfirstlane((int)stride0_elems);

  v4u g0;
  g0[0] = 1u;                                     // count=1 (valid user D#)
  g0[1] = lb;                                     // lds_addr (bytes)
  g0[2] = ga_lo;                                  // global_addr[31:0]
  g0[3] = (ga_hi & 0x01FFFFFFu) | (2u << 30);     // global_addr[56:32] | type=2

  v8i g1;
  g1[0] = (int)(1u << 16);                        // data_size=1 -> 2 bytes/elem
  g1[1] = (int)((td0 & 0xFFFFu) << 16);           // tensor_dim0[15:0]
  g1[2] = (int)(((td0 >> 16) & 0xFFFFu) | ((td1 & 0xFFFFu) << 16));
  g1[3] = (int)(((td1 >> 16) & 0xFFFFu) | ((tl0 & 0xFFFFu) << 16)); // tile_dim0
  g1[4] = (int)(tl1 & 0xFFFFu);                   // tile_dim1 (tile_dim2=0)
  g1[5] = (int)st0;                               // tensor_dim0_stride[31:0]
  g1[6] = 0;                                      // stride0[47:32], dim1_stride lo
  g1[7] = 0;

  v4i z4 = {};
#if __has_include(<hip/amd_detail/amd_gfx1250_TDM.h>)
  v8i z8 = {};
  __builtin_amdgcn_tensor_load_to_lds(g0, g1, z4, z4, z8, 0);   // clang-23 form
#else
  __builtin_amdgcn_tensor_load_to_lds(g0, g1, z4, z4, 0);       // ROCm 7.2 form
#endif
}

// ---------------------------------------------------------------------------
// One 16x32 output tile of C = A @ W^T (two WMMAs share one A fragment).
// A: [M x K] row-major f16, W: [N x K] row-major f16.
// Fragments use contiguous k order (k = 16*(lane/16) + s) in both A and B.
// ---------------------------------------------------------------------------
__device__ __forceinline__ void gemm16x32(const _Float16* __restrict__ A,
                                          const _Float16* __restrict__ W,
                                          int m0, int n0, int K,
                                          v8f& c0, v8f& c1) {
  const int lane = threadIdx.x & 31;
  const int koff = (lane < 16) ? 0 : 16;
  const int kb2  = (lane < 16) ? 0 : 16;
  const _Float16* ap = A + (size_t)(m0 + (lane & 15)) * K + koff;
  const _Float16* w0 = W + (size_t)(n0 + (lane & 15)) * K + kb2;
  const _Float16* w1 = w0 + (size_t)16 * K;
  c0 = {}; c1 = {};
  for (int kk = 0; kk < K; kk += 32) {
    v8h alo = *(const v8h*)(ap + kk);
    v8h ahi = *(const v8h*)(ap + kk + 8);
    v16h a = __builtin_shufflevector(alo, ahi,
                0,1,2,3,4,5,6,7,8,9,10,11,12,13,14,15);
    v16h b0 = *(const v16h*)(w0 + kk);
    v16h b1 = *(const v16h*)(w1 + kk);
    if (kk + 64 < K) {                       // global_prefetch_b8
      __builtin_prefetch(ap + kk + 64, 0, 1);
      __builtin_prefetch(w0 + kk + 64, 0, 1);
      __builtin_prefetch(w1 + kk + 64, 0, 1);
    }
    c0 = __builtin_amdgcn_wmma_f32_16x16x32_f16(false, a, false, b0,
                                                (short)0, c0, false, false);
    c1 = __builtin_amdgcn_wmma_f32_16x16x32_f16(false, a, false, b1,
                                                (short)0, c1, false, false);
  }
}

// ---------------------------------------------------------------------------
// QKV projection + RoPE. grid = (M/16, N/256, 3), block = 256 (8 waves).
// mode: 0=Q(rope)->[B,H,S,DK], 1=K(rope)->[B,H,S,DK], 2=V->[B,H,DK,S] (transposed!)
// ---------------------------------------------------------------------------
__global__ __launch_bounds__(256)
void qkv_rope_kernel(const _Float16* __restrict__ x16,
                     const _Float16* __restrict__ wq,
                     const _Float16* __restrict__ wk,
                     const _Float16* __restrict__ wv,
                     _Float16* __restrict__ qh,
                     _Float16* __restrict__ kh,
                     _Float16* __restrict__ vt) {
  const int wave = threadIdx.x >> 5;
  const int lane = threadIdx.x & 31;
  const int m0 = blockIdx.x * 16;
  const int n0 = blockIdx.y * 256 + wave * 32;
  const int mode = blockIdx.z;

  const _Float16* W = (mode == 0) ? wq : (mode == 1) ? wk : wv;

  v8f c0, c1;
  gemm16x32(x16, W, m0, n0, D_MODEL, c0, c1);

#pragma unroll
  for (int j = 0; j < 2; ++j) {
    const v8f& c = (j == 0) ? c0 : c1;
    const int n = n0 + j * 16 + (lane & 15);
    const int h = n >> 6;
    const int d = n & (DK - 1);
    const float invf =
        __expf(-((float)(d & ~1)) * (9.210340371976184f / (float)DK));
#pragma unroll
    for (int r = 0; r < 8; ++r) {
      const int m = r + ((lane >> 4) << 3);
      const int token = m0 + m;
      const int s = token & (SEQ - 1);
      const int b = token >> 11;
      const float v = c[r];
      const float angle = (float)s * invf;
      const float cs = __cosf(angle), sn = __sinf(angle);
      const float other = __shfl_xor(v, 1, 32);
      const float roped = ((d & 1) == 0) ? (v * cs - other * sn)
                                         : (other * sn + v * cs);
      const int bh = b * NUM_HEADS + h;
      if (mode == 0) {
        qh[(((size_t)bh * SEQ) + s) * DK + d] = (_Float16)roped;
      } else if (mode == 1) {
        kh[(((size_t)bh * SEQ) + s) * DK + d] = (_Float16)roped;
      } else {
        vt[(((size_t)bh * DK) + d) * SEQ + s] = (_Float16)v;   // transposed
      }
    }
  }
}

// ---------------------------------------------------------------------------
// Flash attention: grid = (S/64, B*H), block = 128 (4 waves).
// Each wave owns 16 query rows; per 32-key tile: 4 score WMMAs, online
// softmax (shfl_xor reductions), LDS transpose of P, 4 PV WMMAs.
// K and V^T tiles are DMAed into per-wave LDS double buffers by the TDM;
// s_wait_tensorcnt 2 keeps one tile in flight while the previous computes.
// ---------------------------------------------------------------------------
__global__ __launch_bounds__(128)
void flash_attn_kernel(const _Float16* __restrict__ qh,
                       const _Float16* __restrict__ kh,
                       const _Float16* __restrict__ vt,
                       _Float16* __restrict__ o16) {
  __shared__ __align__(32) _Float16 ldsK[4][2][32 * DK];   // [keys 32][dk 64]
  __shared__ __align__(32) _Float16 ldsV[4][2][DK * 32];   // [dim 64][keys 32]
  __shared__ __align__(32) _Float16 ldsP[4][16][32];       // P staging

  const int wave = threadIdx.x >> 5;
  const int lane = threadIdx.x & 31;
  const int bh = blockIdx.y;
  const int q0 = blockIdx.x * 64 + wave * 16;

  const _Float16* Qp = qh + (size_t)bh * SEQ * DK;
  const _Float16* Kp = kh + (size_t)bh * SEQ * DK;
  const _Float16* Vp = vt + (size_t)bh * DK * SEQ;         // [64][2048]

  const int nb   = lane & 15;
  const int koff = (lane < 16) ? 0 : 16;
  const int kb2  = (lane < 16) ? 0 : 16;

  // Q fragments (A layout) for dk 0..31 / 32..63, resident for the whole loop.
  const _Float16* qp = Qp + (size_t)(q0 + nb) * DK + koff;
  v8h q00 = *(const v8h*)(qp);
  v8h q01 = *(const v8h*)(qp + 8);
  v8h q10 = *(const v8h*)(qp + 32);
  v8h q11 = *(const v8h*)(qp + 40);
  v16h aq0 = __builtin_shufflevector(q00, q01, 0,1,2,3,4,5,6,7,8,9,10,11,12,13,14,15);
  v16h aq1 = __builtin_shufflevector(q10, q11, 0,1,2,3,4,5,6,7,8,9,10,11,12,13,14,15);

  v8f acc0 = {}, acc1 = {}, acc2 = {}, acc3 = {};
  float rowmax[8], rowsum[8];
#pragma unroll
  for (int r = 0; r < 8; ++r) { rowmax[r] = NEG_BIG; rowsum[r] = 0.0f; }

  const int kbmax = (q0 + 15) >> 5;

  // Stage key-tile kb into per-wave LDS buffer `buf` via the TDM.
  auto stage = [&](int kb, int buf) {
    const int kbase = kb << 5;
    // K tile: 32 consecutive rows of [SEQ][64] -> LDS [32][64]
    tdm_load_2d((unsigned)(uintptr_t)&ldsK[wave][buf][0],
                Kp + (size_t)kbase * DK,
                /*tensor_d0=*/DK, /*tensor_d1=*/SEQ,
                /*tile_d0=*/DK, /*tile_d1=*/32, /*stride0=*/DK);
    // V^T tile: 64 rows x 32 cols of [64][SEQ] -> LDS [64][32]
    tdm_load_2d((unsigned)(uintptr_t)&ldsV[wave][buf][0],
                Vp + kbase,
                /*tensor_d0=*/SEQ, /*tensor_d1=*/DK,
                /*tile_d0=*/32, /*tile_d1=*/DK, /*stride0=*/SEQ);
  };

  stage(0, 0);
  for (int kb = 0; kb <= kbmax; ++kb) {
    const int kbase = kb << 5;
    const int buf = kb & 1;
    const bool more = (kb < kbmax);
    if (more) {
      stage(kb + 1, buf ^ 1);                       // prefetch next tile
      __builtin_amdgcn_s_wait_tensorcnt((short)2);  // current tile resident
    } else {
      __builtin_amdgcn_s_wait_tensorcnt((short)0);
    }

    // ---- scores: two 16x16 tiles from LDS K buffer ----
    v8f sc[2];
#pragma unroll
    for (int t = 0; t < 2; ++t) {
      const _Float16* kp0 = &ldsK[wave][buf][(t * 16 + nb) * DK + kb2];
      v16h bk0 = *(const v16h*)(kp0);        // dk 0..31 slice
      v16h bk1 = *(const v16h*)(kp0 + 32);   // dk 32..63 slice
      v8f cc = {};
      cc = __builtin_amdgcn_wmma_f32_16x16x32_f16(false, aq0, false, bk0,
                                                  (short)0, cc, false, false);
      cc = __builtin_amdgcn_wmma_f32_16x16x32_f16(false, aq1, false, bk1,
                                                  (short)0, cc, false, false);
      sc[t] = cc;
    }

    // ---- online softmax update ----
#pragma unroll
    for (int r = 0; r < 8; ++r) {
      const int m = r + ((lane >> 4) << 3);
      const int qrow = q0 + m;
      float s0 = sc[0][r] * 0.125f;                  // 1/sqrt(64)
      float s1 = sc[1][r] * 0.125f;
      if (kbase + nb > qrow)      s0 = NEG_BIG;      // causal mask
      if (kbase + 16 + nb > qrow) s1 = NEG_BIG;
      float tm = fmaxf(s0, s1);
#pragma unroll
      for (int off = 1; off < 16; off <<= 1)
        tm = fmaxf(tm, __shfl_xor(tm, off, 32));
      const float mnew = fmaxf(rowmax[r], tm);
      const float p0 = __expf(s0 - mnew);
      const float p1 = __expf(s1 - mnew);
      float ts = p0 + p1;
#pragma unroll
      for (int off = 1; off < 16; off <<= 1)
        ts += __shfl_xor(ts, off, 32);
      const float f = __expf(rowmax[r] - mnew);
      rowmax[r] = mnew;
      rowsum[r] = rowsum[r] * f + ts;
      acc0[r] *= f; acc1[r] *= f; acc2[r] *= f; acc3[r] *= f;
      ldsP[wave][m][nb]      = (_Float16)p0;
      ldsP[wave][m][16 + nb] = (_Float16)p1;
    }

    // ---- reload P as an A-fragment (16 rows x 32 keys) ----
    v16h pf;
    {
      const int prow = lane & 15;
#pragma unroll
      for (int e = 0; e < 8; ++e) pf[e]     = ldsP[wave][prow][koff + e];
#pragma unroll
      for (int e = 0; e < 8; ++e) pf[8 + e] = ldsP[wave][prow][koff + 8 + e];
    }

    // ---- PV: out[16x64] += P[16x32] @ V[32x64]; B-frags contiguous in LDS ----
#pragma unroll
    for (int j = 0; j < 4; ++j) {
      v16h bv = *(const v16h*)&ldsV[wave][buf][(j * 16 + nb) * 32 + kb2];
      v8f& a = (j == 0) ? acc0 : (j == 1) ? acc1 : (j == 2) ? acc2 : acc3;
      a = __builtin_amdgcn_wmma_f32_16x16x32_f16(false, pf, false, bv,
                                                 (short)0, a, false, false);
    }
  }

  // ---- normalize and scatter to [B*S, D_MODEL] ----
  const int b = bh >> 4;
  const int h = bh & (NUM_HEADS - 1);
#pragma unroll
  for (int j = 0; j < 4; ++j) {
    const v8f& a = (j == 0) ? acc0 : (j == 1) ? acc1 : (j == 2) ? acc2 : acc3;
#pragma unroll
    for (int r = 0; r < 8; ++r) {
      const int m = r + ((lane >> 4) << 3);
      const size_t token = (size_t)b * SEQ + q0 + m;
      const int dim = h * DK + j * 16 + nb;
      o16[token * D_MODEL + dim] = (_Float16)(a[r] / rowsum[r]);
    }
  }
}

// ---------------------------------------------------------------------------
// Output projection: out[f32] = O16 @ Wo^T. grid = (M/16, N/256), block = 256.
// ---------------------------------------------------------------------------
__global__ __launch_bounds__(256)
void out_proj_kernel(const _Float16* __restrict__ o16,
                     const _Float16* __restrict__ wo,
                     float* __restrict__ out) {
  const int wave = threadIdx.x >> 5;
  const int lane = threadIdx.x & 31;
  const int m0 = blockIdx.x * 16;
  const int n0 = blockIdx.y * 256 + wave * 32;
  v8f c0, c1;
  gemm16x32(o16, wo, m0, n0, D_MODEL, c0, c1);
#pragma unroll
  for (int j = 0; j < 2; ++j) {
    const v8f& c = (j == 0) ? c0 : c1;
    const int n = n0 + j * 16 + (lane & 15);
#pragma unroll
    for (int r = 0; r < 8; ++r) {
      const int m = r + ((lane >> 4) << 3);
      out[(size_t)(m0 + m) * D_MODEL + n] = c[r];
    }
  }
}

// ---------------------------------------------------------------------------
// Host launcher. Workspace (f16, ~48 MB):
//   x16[4096*1024] | wq|wk|wv|wo [1024^2 each] | q16,k16 [B,H,S,DK]
//   | vt16 [B,H,DK,S] | o16 [4096*1024]
// ---------------------------------------------------------------------------
extern "C" void kernel_launch(void* const* d_in, const int* in_sizes, int n_in,
                              void* d_out, int out_size, void* d_ws, size_t ws_size,
                              hipStream_t stream) {
  const float* x  = (const float*)d_in[0];
  const float* Wq = (const float*)d_in[1];
  const float* Wk = (const float*)d_in[2];
  const float* Wv = (const float*)d_in[3];
  const float* Wo = (const float*)d_in[4];
  float* out = (float*)d_out;

  const size_t NX = (size_t)MTOT * D_MODEL;       // 4,194,304
  const size_t NW = (size_t)D_MODEL * D_MODEL;    // 1,048,576

  _Float16* ws   = (_Float16*)d_ws;
  _Float16* x16  = ws;
  _Float16* wq16 = x16  + NX;
  _Float16* wk16 = wq16 + NW;
  _Float16* wv16 = wk16 + NW;
  _Float16* wo16 = wv16 + NW;
  _Float16* q16  = wo16 + NW;
  _Float16* k16  = q16  + NX;
  _Float16* vt16 = k16  + NX;
  _Float16* o16  = vt16 + NX;

  cvt_f32_to_f16<<<2048, 256, 0, stream>>>(x,  x16,  (int)NX);
  cvt_f32_to_f16<<<1024, 256, 0, stream>>>(Wq, wq16, (int)NW);
  cvt_f32_to_f16<<<1024, 256, 0, stream>>>(Wk, wk16, (int)NW);
  cvt_f32_to_f16<<<1024, 256, 0, stream>>>(Wv, wv16, (int)NW);
  cvt_f32_to_f16<<<1024, 256, 0, stream>>>(Wo, wo16, (int)NW);

  qkv_rope_kernel<<<dim3(MTOT / 16, D_MODEL / 256, 3), 256, 0, stream>>>(
      x16, wq16, wk16, wv16, q16, k16, vt16);

  flash_attn_kernel<<<dim3(SEQ / 64, BATCH * NUM_HEADS), 128, 0, stream>>>(
      q16, k16, vt16, o16);

  out_proj_kernel<<<dim3(MTOT / 16, D_MODEL / 256), 256, 0, stream>>>(
      o16, wo16, out);
}
